// TritonSSMStep_25451976196326
// MI455X (gfx1250) — matrix operand order repaired and checked by reading, not compiled, not run on hardware
//
#include <hip/hip_runtime.h>
#include <hip/hip_bf16.h>

// Problem constants
#define D_   1024
#define N_   64
#define R_   4
#define B_   4
#define L_   4096
#define T_   (B_ * L_)     // 16384 tokens
#define NC_  384           // padded fused projection channels (356 real, pad to 384)
#define GC_  256           // R*N, second-GEMM K

typedef __attribute__((ext_vector_type(16))) __bf16 v16bf;
typedef __attribute__((ext_vector_type(8)))  __bf16 v8bf;
typedef __attribute__((ext_vector_type(8)))  float  v8f;

// Workspace layout (bytes)
#define XB_OFF    ((size_t)0)                         // T_*D_ bf16      = 32 MB
#define WCAT_OFF  (XB_OFF   + (size_t)T_ * D_ * 2)    // NC_*D_ bf16     = 768 KB
#define WOUT_OFF  (WCAT_OFF + (size_t)NC_ * D_ * 2)   // D_*GC_ bf16     = 512 KB
#define BCAT_OFF  (WOUT_OFF + (size_t)D_ * GC_ * 2)   // NC_ f32
#define P_OFF     (BCAT_OFF + (size_t)NC_ * 4)        // T_*NC_ f32      = 24 MB
#define G_OFF     (P_OFF    + (size_t)T_ * NC_ * 4)   // T_*GC_ bf16     = 8 MB

__device__ __forceinline__ float sigmoidf_(float z) { return 1.0f / (1.0f + __expf(-z)); }
__device__ __forceinline__ float siluf_(float z)    { return z * sigmoidf_(z); }

// ---------------------------------------------------------------------------
// Pack [W_B;W_X;W_decay;W_theta] -> Wcat bf16 (384 x 1024), biases -> bcat f32
// ---------------------------------------------------------------------------
__global__ void pack_wcat(const float* __restrict__ W_B,  const float* __restrict__ b_B,
                          const float* __restrict__ W_X,  const float* __restrict__ b_X,
                          const float* __restrict__ W_dc, const float* __restrict__ b_dc,
                          const float* __restrict__ W_th, const float* __restrict__ b_th,
                          __bf16* __restrict__ wcat, float* __restrict__ bcat) {
    int idx = blockIdx.x * blockDim.x + threadIdx.x;      // over NC_*D_
    int row = idx / D_, col = idx % D_;
    float w;
    if      (row < 256) w = W_B [(row      ) * D_ + col];
    else if (row < 260) w = W_X [(row - 256) * D_ + col];
    else if (row < 324) w = W_dc[(row - 260) * D_ + col];
    else if (row < 356) w = W_th[(row - 324) * D_ + col];
    else                w = 0.0f;
    wcat[idx] = (__bf16)w;
    if (idx < NC_) {
        float bb;
        if      (idx < 256) bb = b_B [idx];
        else if (idx < 260) bb = b_X [idx - 256];
        else if (idx < 324) bb = b_dc[idx - 260];
        else if (idx < 356) bb = b_th[idx - 324];
        else                bb = 0.0f;
        bcat[idx] = bb;
    }
}

__global__ void pack_wout(const float* __restrict__ W_out, __bf16* __restrict__ wout) {
    int idx = blockIdx.x * blockDim.x + threadIdx.x;      // over D_*GC_
    wout[idx] = (__bf16)W_out[idx];
}

__global__ void convert_x(const float* __restrict__ x, __bf16* __restrict__ xb) {
    int idx = blockIdx.x * blockDim.x + threadIdx.x;      // one thread = 8 elems
    const float* s = x + (size_t)idx * 8;
    v8bf o;
#pragma unroll
    for (int i = 0; i < 8; ++i) o[i] = (__bf16)s[i];
    *(v8bf*)(xb + (size_t)idx * 8) = o;
}

// ---------------------------------------------------------------------------
// GEMM1: P[t][c] = xb[t][:] . wcat[c][:] + bcat[c]   (t:16384, c:384, K:1024)
// Each wave computes a 16x64 strip (4 accumulators) so one A fragment feeds
// 4 v_wmma_f32_16x16x32_bf16 ops. block = 192 (6 waves), grid = (1024, 1);
// wave w owns N-tiles [4w, 4w+4) -> covers all 24 column tiles.
// ---------------------------------------------------------------------------
__global__ void gemm1(const __bf16* __restrict__ xb, const __bf16* __restrict__ wcat,
                      const float* __restrict__ bcat, float* __restrict__ P) {
    const int lane = threadIdx.x & 31;
    const int wv   = threadIdx.x >> 5;                    // 0..5
    const int mt   = blockIdx.x;
    const int nt0  = wv * 4;                              // first of 4 N-tiles
    const int half = lane >> 4;
    const int lm   = lane & 15;

    const __bf16* arow = xb + (size_t)(mt * 16 + lm) * D_;
    const __bf16* brow0 = wcat + (size_t)((nt0 + 0) * 16 + lm) * D_;
    const __bf16* brow1 = wcat + (size_t)((nt0 + 1) * 16 + lm) * D_;
    const __bf16* brow2 = wcat + (size_t)((nt0 + 2) * 16 + lm) * D_;
    const __bf16* brow3 = wcat + (size_t)((nt0 + 3) * 16 + lm) * D_;

    v8f acc0{}, acc1{}, acc2{}, acc3{};
    for (int kk = 0; kk < D_; kk += 32) {
        // A fragment (16x32, row-major): lane<16 -> K {0..15}, lane>=16 -> {16..31}
        v16bf a  = *(const v16bf*)(arow + kk + half * 16);
        // B fragments (32x16): column n = lm, K = 16*half + e -> one contiguous 32B load each
        v16bf b0 = *(const v16bf*)(brow0 + kk + half * 16);
        v16bf b1 = *(const v16bf*)(brow1 + kk + half * 16);
        v16bf b2 = *(const v16bf*)(brow2 + kk + half * 16);
        v16bf b3 = *(const v16bf*)(brow3 + kk + half * 16);
        acc0 = __builtin_amdgcn_wmma_f32_16x16x32_bf16(false, a, false, b0, (short)0, acc0, false, false);
        acc1 = __builtin_amdgcn_wmma_f32_16x16x32_bf16(false, a, false, b1, (short)0, acc1, false, false);
        acc2 = __builtin_amdgcn_wmma_f32_16x16x32_bf16(false, a, false, b2, (short)0, acc2, false, false);
        acc3 = __builtin_amdgcn_wmma_f32_16x16x32_bf16(false, a, false, b3, (short)0, acc3, false, false);
    }
#pragma unroll
    for (int j = 0; j < 4; ++j) {
        const v8f& acc = (j == 0) ? acc0 : (j == 1) ? acc1 : (j == 2) ? acc2 : acc3;
        const int  col = (nt0 + j) * 16 + lm;
        const float bias = bcat[col];
#pragma unroll
        for (int v = 0; v < 8; ++v) {
            int row = mt * 16 + v + 8 * half;             // C/D layout: M = v + 8*half, N = lm
            P[(size_t)row * NC_ + col] = acc[v] + bias;
        }
    }
}

// ---------------------------------------------------------------------------
// Elementwise: rotation + decay gate. One block (128 thr) per token.
// thread -> (r = tid>>5, k = tid&31) handles the (2k, 2k+1) channel pair.
// Writes H_new (d_out part 1, layout (B,R,L,N)) and G bf16 (channel r*N+n).
// ---------------------------------------------------------------------------
__global__ void elementwise(const float* __restrict__ P, const float* __restrict__ H,
                            const int* __restrict__ lidx, float* __restrict__ Hnew,
                            __bf16* __restrict__ G) {
    const int t   = blockIdx.x;
    const int r   = threadIdx.x >> 5;
    const int k   = threadIdx.x & 31;
    const int b   = t / L_;
    const int l   = t % L_;
    const float mult = (float)(*lidx + 1);

    const float* p = P + (size_t)t * NC_;
    const int ne = 2 * k, no = ne + 1;

    float s1 = siluf_(p[ne * R_ + r]);                    // B_proj even
    float s2 = siluf_(p[no * R_ + r]);                    // B_proj odd
    float xa = siluf_(p[256 + r]);                        // X_r
    float de = sigmoidf_(p[260 + ne]);                    // decay even
    float dn = sigmoidf_(p[260 + no]);                    // decay odd
    float th = p[324 + k] * mult;
    float c  = __cosf(th), s = __sinf(th);

    float bre = s1 * c - s2 * s;
    float bro = s1 * s + s2 * c;

    size_t hi = (((size_t)b * R_ + r) * L_ + l) * N_ + ne;
    float he = de * H[hi]     + bre * xa;
    float ho = dn * H[hi + 1] + bro * xa;
    Hnew[hi]     = he;
    Hnew[hi + 1] = ho;

    __bf16* g = G + (size_t)t * GC_ + r * N_;
    g[ne] = (__bf16)he;
    g[no] = (__bf16)ho;
}

// ---------------------------------------------------------------------------
// GEMM2: out[t][d] = silu(G[t][:] . W_out[d][:] + b_out[d])  (K = 256)
// 4 N-tiles per wave; block = 256 (8 waves), grid = (1024, 2);
// wave-job (gy*8 + w) owns N-tiles [4*job, 4*job+4) -> covers all 64 tiles.
// ---------------------------------------------------------------------------
__global__ void gemm2(const __bf16* __restrict__ G, const __bf16* __restrict__ wout,
                      const float* __restrict__ b_out, float* __restrict__ out2) {
    const int lane = threadIdx.x & 31;
    const int wv   = threadIdx.x >> 5;
    const int mt   = blockIdx.x;
    const int nt0  = (blockIdx.y * 8 + wv) * 4;           // 0,4,...,60
    const int half = lane >> 4;
    const int lm   = lane & 15;

    const __bf16* arow = G + (size_t)(mt * 16 + lm) * GC_;
    const __bf16* brow0 = wout + (size_t)((nt0 + 0) * 16 + lm) * GC_;
    const __bf16* brow1 = wout + (size_t)((nt0 + 1) * 16 + lm) * GC_;
    const __bf16* brow2 = wout + (size_t)((nt0 + 2) * 16 + lm) * GC_;
    const __bf16* brow3 = wout + (size_t)((nt0 + 3) * 16 + lm) * GC_;

    v8f acc0{}, acc1{}, acc2{}, acc3{};
#pragma unroll
    for (int kk = 0; kk < GC_; kk += 32) {
        v16bf a  = *(const v16bf*)(arow + kk + half * 16);
        v16bf b0 = *(const v16bf*)(brow0 + kk + half * 16);
        v16bf b1 = *(const v16bf*)(brow1 + kk + half * 16);
        v16bf b2 = *(const v16bf*)(brow2 + kk + half * 16);
        v16bf b3 = *(const v16bf*)(brow3 + kk + half * 16);
        acc0 = __builtin_amdgcn_wmma_f32_16x16x32_bf16(false, a, false, b0, (short)0, acc0, false, false);
        acc1 = __builtin_amdgcn_wmma_f32_16x16x32_bf16(false, a, false, b1, (short)0, acc1, false, false);
        acc2 = __builtin_amdgcn_wmma_f32_16x16x32_bf16(false, a, false, b2, (short)0, acc2, false, false);
        acc3 = __builtin_amdgcn_wmma_f32_16x16x32_bf16(false, a, false, b3, (short)0, acc3, false, false);
    }
#pragma unroll
    for (int j = 0; j < 4; ++j) {
        const v8f& acc = (j == 0) ? acc0 : (j == 1) ? acc1 : (j == 2) ? acc2 : acc3;
        const int  dcol = (nt0 + j) * 16 + lm;
        const float bias = b_out[dcol];
#pragma unroll
        for (int v = 0; v < 8; ++v) {
            float z = acc[v] + bias;
            float o = z * (1.0f / (1.0f + __expf(-z)));
            int row = mt * 16 + v + 8 * half;
            out2[(size_t)row * D_ + dcol] = o;
        }
    }
}

// ---------------------------------------------------------------------------
extern "C" void kernel_launch(void* const* d_in, const int* in_sizes, int n_in,
                              void* d_out, int out_size, void* d_ws, size_t ws_size,
                              hipStream_t stream) {
    const float* x      = (const float*)d_in[0];
    const float* H      = (const float*)d_in[1];
    const float* W_B    = (const float*)d_in[2];
    const float* b_B    = (const float*)d_in[3];
    const float* W_X    = (const float*)d_in[4];
    const float* b_X    = (const float*)d_in[5];
    const float* W_dc   = (const float*)d_in[6];
    const float* b_dc   = (const float*)d_in[7];
    const float* W_th   = (const float*)d_in[8];
    const float* b_th   = (const float*)d_in[9];
    const float* W_out  = (const float*)d_in[10];
    const float* b_out  = (const float*)d_in[11];
    const int*   lidx   = (const int*)d_in[12];

    char* ws = (char*)d_ws;
    __bf16* xb   = (__bf16*)(ws + XB_OFF);
    __bf16* wcat = (__bf16*)(ws + WCAT_OFF);
    __bf16* wout = (__bf16*)(ws + WOUT_OFF);
    float*  bcat = (float*)(ws + BCAT_OFF);
    float*  P    = (float*)(ws + P_OFF);
    __bf16* G    = (__bf16*)(ws + G_OFF);

    float* Hnew = (float*)d_out;                          // (B,R,L,N) = 4,194,304 f32
    float* out2 = (float*)d_out + (size_t)B_ * R_ * L_ * N_;

    pack_wcat<<<dim3((NC_ * D_) / 256), dim3(256), 0, stream>>>(
        W_B, b_B, W_X, b_X, W_dc, b_dc, W_th, b_th, wcat, bcat);
    pack_wout<<<dim3((D_ * GC_) / 256), dim3(256), 0, stream>>>(W_out, wout);
    convert_x<<<dim3((size_t)T_ * D_ / (8 * 256)), dim3(256), 0, stream>>>(x, xb);

    gemm1<<<dim3(T_ / 16), dim3(192), 0, stream>>>(xb, wcat, bcat, P);
    elementwise<<<dim3(T_), dim3(128), 0, stream>>>(P, H, lidx, Hnew, G);
    gemm2<<<dim3(T_ / 16, 2), dim3(256), 0, stream>>>(G, wout, b_out, out2);
}
